// LSTMDecoder_30872224924301
// MI455X (gfx1250) — matrix operand order repaired and checked by reading, not compiled, not run on hardware
//
#include <hip/hip_runtime.h>

// ---------------------------------------------------------------------------
// 2-layer LSTM decoder for MI455X (gfx1250, wave32, WMMA), fully fused cells.
// B=256, D=512, T=64.  Per step/layer ONE kernel:
//   K-split x4 across waves (2048 waves/kernel for latency hiding),
//   gates(i,f,g,o) via v_wmma_f32_16x16x32_bf16 with a sched_barrier-enforced
//   software pipeline (loads of slice k+1 issue before WMMAs of slice k),
//   LDS cross-wave reduction, then fused sigmoid/tanh cell update.
// Weights converted once to bf16 (8 MB, L2-resident).
// ---------------------------------------------------------------------------

typedef __bf16 bf16;
typedef __attribute__((ext_vector_type(16))) __bf16 v16bf;
typedef __attribute__((ext_vector_type(8)))  __bf16 v8bf;
typedef __attribute__((ext_vector_type(8)))  float  v8f;

#define BATCH  256
#define DIM    512
#define GDIM   2048   // 4*DIM
#define TSTEPS 64
#define KSPLIT 256    // K elements per wave (of 1024 fused)

#if defined(__has_builtin)
#if __has_builtin(__builtin_amdgcn_sched_barrier)
#define SCHED_FENCE() __builtin_amdgcn_sched_barrier(0)
#endif
#endif
#ifndef SCHED_FENCE
#define SCHED_FENCE()
#endif

// ---------------- WMMA fragment helpers (contiguous K = 16h + s, A and B alike) ----------

// A fragment 16x32 bf16: lanes 0-15 -> rows M=0..15 with K in {0..15},
// lanes 16-31 -> same rows with K in {16..31}. Two 16B loads per lane.
__device__ __forceinline__ v16bf load_a_frag(const bf16* p) {
  v8bf lo = *(const v8bf*)(p);
  v8bf hi = *(const v8bf*)(p + 8);
  v16bf r;
#pragma unroll
  for (int i = 0; i < 8; ++i) { r[i] = lo[i]; r[i + 8] = hi[i]; }
  return r;
}

#define WMMA_BF16(a, b, c) \
  __builtin_amdgcn_wmma_f32_16x16x32_bf16(false, (a), false, (b), (short)0, \
                                          (c), false, false)

// Fragment bundle for one K=32 slice: A + four gate B fragments.
struct Frags {
  v16bf a, bi, bf, bg, bo;
};

__device__ __forceinline__ Frags load_frags(const bf16* __restrict__ aBase,
                                            const bf16* __restrict__ wi,
                                            const bf16* __restrict__ wf,
                                            const bf16* __restrict__ wg,
                                            const bf16* __restrict__ wo,
                                            int k) {
  Frags f;
  f.a  = load_a_frag(aBase + k);
  f.bi = *(const v16bf*)(wi + k);
  f.bf = *(const v16bf*)(wf + k);
  f.bg = *(const v16bf*)(wg + k);
  f.bo = *(const v16bf*)(wo + k);
  return f;
}

__device__ __forceinline__ void mma_frags(const Frags& f, v8f& ci, v8f& cf,
                                          v8f& cg, v8f& co) {
  ci = WMMA_BF16(f.a, f.bi, ci);
  cf = WMMA_BF16(f.a, f.bf, cf);
  cg = WMMA_BF16(f.a, f.bg, cg);
  co = WMMA_BF16(f.a, f.bo, co);
}

__device__ __forceinline__ float sigf(float x) {
  return 1.0f / (1.0f + __expf(-x));
}

// ---------------- fused LSTM layer kernel -----------------------------------
// Block: 256 threads = 8 waves = 2 (M) x 4 (K-quarter); tile 32(M) x 16(d).
// Each wave: 16x16 (row,d) tile, 4 accumulators = gates i,f,g,o, K=256.
// K-quarters: {x[0:256), x[256:512), h[0:256), h[256:512)}.
// Cross-wave reduction via LDS; lead waves (q==0) run the fused cell.
// Grid: (BATCH/32, DIM/16) = (8, 32) = 256 blocks -> 2048 waves.
__global__ __launch_bounds__(256) void lstm_layer_fused(
    const bf16* __restrict__ A0,   // x input  [256,512] bf16
    const bf16* __restrict__ A1,   // h input  [256,512] bf16
    const bf16* __restrict__ W0,   // W_ih bf16 [2048,512] row-major
    const bf16* __restrict__ W1,   // W_hh bf16 [2048,512] row-major
    const float* __restrict__ bias,// b_ih + b_hh  [2048]
    float* __restrict__ c,         // cell state [256,512] (in/out)
    bf16*  __restrict__ h_bf,      // bf16 h for next GEMMs (out)
    float* __restrict__ h_out)     // f32 output slice (layer 1) or null
{
  __shared__ float red[8 * 1024];  // 32 KB: [wave][acc 0..3][lane*8 + v]

  const int lane = threadIdx.x & 31;
  const int wave = threadIdx.x >> 5;  // 0..7
  const int q    = wave & 3;          // K-quarter
  const int mpos = wave >> 2;         // 0..1
  const int m0   = blockIdx.x * 32 + mpos * 16;
  const int d0   = blockIdx.y * 16;

  const int l15 = lane & 15;
  const int lhi = lane >> 4;

  // Quarter selects panel (x vs h) and K offset within the 512-wide panel.
  const bf16* A = (q < 2) ? A0 : A1;
  const bf16* W = (q < 2) ? W0 : W1;
  const int kbase = (q & 1) * KSPLIT;

  // A fragment base (row m0+l15, K-half by lane group)
  const bf16* ap = A + (m0 + l15) * DIM + kbase + lhi * 16;

  // B fragment bases: gate g's columns are W rows g*DIM + d  (gates = A@W^T)
  const int wrow = d0 + l15;
  const int kofs = kbase + lhi * 16;
  const bf16* wi = W + (0 * DIM + wrow) * DIM + kofs;
  const bf16* wf = W + (1 * DIM + wrow) * DIM + kofs;
  const bf16* wg = W + (2 * DIM + wrow) * DIM + kofs;
  const bf16* wo = W + (3 * DIM + wrow) * DIM + kofs;

  v8f ai = {}, af_ = {}, ag = {}, ao = {};

  // K=256 per wave: 8 slices, distance-1 software pipeline.
  // SCHED_FENCE() forbids the backend scheduler from sinking slice k+1's
  // loads below slice k's WMMAs (which would recreate s_wait_loadcnt 0
  // before every WMMA); both fragment sets stay live in distinct VGPRs.
  Frags cur = load_frags(ap, wi, wf, wg, wo, 0);
#pragma unroll
  for (int k = 32; k < KSPLIT; k += 32) {
    Frags nxt = load_frags(ap, wi, wf, wg, wo, k);  // issue next loads
    SCHED_FENCE();
    mma_frags(cur, ai, af_, ag, ao);                // consume current
    cur = nxt;
  }
  SCHED_FENCE();
  mma_frags(cur, ai, af_, ag, ao);

  // ---- cross-wave K reduction through LDS ----
  if (q != 0) {
    float* dst = &red[wave * 1024 + lane * 8];
    *(v8f*)(dst +   0) = ai;
    *(v8f*)(dst + 256) = af_;
    *(v8f*)(dst + 512) = ag;
    *(v8f*)(dst + 768) = ao;
  }
  __syncthreads();

  if (q == 0) {
#pragma unroll
    for (int p = 1; p < 4; ++p) {
      const float* src = &red[(wave + p) * 1024 + lane * 8];
      ai  += *(const v8f*)(src +   0);
      af_ += *(const v8f*)(src + 256);
      ag  += *(const v8f*)(src + 512);
      ao  += *(const v8f*)(src + 768);
    }

    const float bi = bias[0 * DIM + wrow];
    const float bf = bias[1 * DIM + wrow];
    const float bg = bias[2 * DIM + wrow];
    const float bo = bias[3 * DIM + wrow];

    // C/D layout: vgpr v, lane<16 -> (M=v,N=lane); lane>=16 -> (M=8+v,N=l-16).
    // Each lane-element holds all four gates of cell (row, d): fuse the cell.
    const int row0 = m0 + lhi * 8;
#pragma unroll
    for (int v = 0; v < 8; ++v) {
      const int idx = (row0 + v) * DIM + d0 + l15;
      const float gi = ai[v] + bi;
      const float gf = af_[v] + bf;
      const float gg = ag[v] + bg;
      const float go = ao[v] + bo;
      const float cn = sigf(gf) * c[idx] + sigf(gi) * tanhf(gg);
      c[idx] = cn;
      const float hn = sigf(go) * tanhf(cn);
      h_bf[idx] = (bf16)hn;
      if (h_out) h_out[idx] = hn;
    }
  }
}

// ---------------- small utility kernels -------------------------------------
__global__ void f32_to_bf16(const float* __restrict__ s, bf16* __restrict__ d,
                            int n) {
  for (int i = blockIdx.x * blockDim.x + threadIdx.x; i < n;
       i += gridDim.x * blockDim.x)
    d[i] = (bf16)s[i];
}

__global__ void add_bias_vec(const float* __restrict__ a,
                             const float* __restrict__ b,
                             float* __restrict__ o, int n) {
  const int i = blockIdx.x * blockDim.x + threadIdx.x;
  if (i < n) o[i] = a[i] + b[i];
}

__global__ void zero_u32(unsigned int* __restrict__ p, int n) {
  for (int i = blockIdx.x * blockDim.x + threadIdx.x; i < n;
       i += gridDim.x * blockDim.x)
    p[i] = 0u;
}

// ---------------- host side --------------------------------------------------
extern "C" void kernel_launch(void* const* d_in, const int* in_sizes, int n_in,
                              void* d_out, int out_size, void* d_ws,
                              size_t ws_size, hipStream_t stream) {
  const float* h_in = (const float*)d_in[0];   // [256,512] initial input
  // d_in[1] is T (== 64, compile-time constant here)
  const float* Wih0 = (const float*)d_in[2];
  const float* Whh0 = (const float*)d_in[3];
  const float* bih0 = (const float*)d_in[4];
  const float* bhh0 = (const float*)d_in[5];
  const float* Wih1 = (const float*)d_in[6];
  const float* Whh1 = (const float*)d_in[7];
  const float* bih1 = (const float*)d_in[8];
  const float* bhh1 = (const float*)d_in[9];
  float* out = (float*)d_out;                  // [64,256,512]

  // ---- workspace layout (bytes) ----
  char* ws = (char*)d_ws;
  const size_t WELEMS = (size_t)GDIM * DIM;          // 1048576 per matrix
  bf16* Wb_ih0 = (bf16*)(ws + 0);                    // 2 MB each
  bf16* Wb_hh0 = (bf16*)(ws + 2097152);
  bf16* Wb_ih1 = (bf16*)(ws + 4194304);
  bf16* Wb_hh1 = (bf16*)(ws + 6291456);
  float* bias0 = (float*)(ws + 8388608);             // 8 KB
  float* bias1 = (float*)(ws + 8396800);             // 8 KB
  bf16* xb     = (bf16*)(ws + 8404992);              // 256 KB
  bf16* h0b    = (bf16*)(ws + 8667136);              // 256 KB  } zeroed
  bf16* h1b    = (bf16*)(ws + 8929280);              // 256 KB  } as one
  float* c0    = (float*)(ws + 9191424);             // 512 KB  } contiguous
  float* c1    = (float*)(ws + 9715712);             // 512 KB  } region

  // ---- one-time-per-launch prep (deterministic, no cross-call state) ----
  const int cvtGrid = 1024;
  f32_to_bf16<<<cvtGrid, 256, 0, stream>>>(Wih0, Wb_ih0, (int)WELEMS);
  f32_to_bf16<<<cvtGrid, 256, 0, stream>>>(Whh0, Wb_hh0, (int)WELEMS);
  f32_to_bf16<<<cvtGrid, 256, 0, stream>>>(Wih1, Wb_ih1, (int)WELEMS);
  f32_to_bf16<<<cvtGrid, 256, 0, stream>>>(Whh1, Wb_hh1, (int)WELEMS);
  f32_to_bf16<<<256, 256, 0, stream>>>(h_in, xb, BATCH * DIM);
  add_bias_vec<<<8, 256, 0, stream>>>(bih0, bhh0, bias0, GDIM);
  add_bias_vec<<<8, 256, 0, stream>>>(bih1, bhh1, bias1, GDIM);
  // zero h0b,h1b,c0,c1 : 2*256KB + 2*512KB = 1536 KB = 393216 dwords
  zero_u32<<<768, 256, 0, stream>>>((unsigned int*)h0b, 393216);

  // ---- sequential LSTM chain: 2 fused kernels per step ----
  const dim3 grid(BATCH / 32, DIM / 16);  // (8, 32) = 256 blocks

  for (int t = 0; t < TSTEPS; ++t) {
    const bf16* x = (t == 0) ? xb : h1b;  // layer-1 output feeds back
    // layer 0 (no f32 output)
    lstm_layer_fused<<<grid, 256, 0, stream>>>(x, h0b, Wb_ih0, Wb_hh0, bias0,
                                               c0, h0b, nullptr);
    // layer 1 (writes output slice t)
    lstm_layer_fused<<<grid, 256, 0, stream>>>(h0b, h1b, Wb_ih1, Wb_hh1, bias1,
                                               c1, h1b,
                                               out + (size_t)t * BATCH * DIM);
  }
}
